// MultiHeadAttention_50414326120853
// MI455X (gfx1250) — compile-verified, matrix operand order checked by reading, not run on hardware
//
#include <hip/hip_runtime.h>

// ---------------- problem constants ----------------
#define H_   16
#define B_   2
#define L_   2048
#define DM_  1024
#define DK_  64
#define DV_  64
#define NP_  1024               // H*DK == H*DV == DM
#define ANGC_ (-0.0179889460f)  // -2/1024 * ln(10000)

typedef __attribute__((ext_vector_type(16))) _Float16 v16h;
typedef __attribute__((ext_vector_type(8)))  _Float16 v8h;
typedef __attribute__((ext_vector_type(4)))  _Float16 v4h;
typedef __attribute__((ext_vector_type(8)))  float    v8f;
typedef __attribute__((ext_vector_type(4)))  float    v4f;

// ---- fragment helpers (ISA 7.12.2 layouts) --------------------------------
// A/B fragment of v_wmma_f32_16x16x32_f16 from a K-contiguous f16 row:
// lane<16 -> K {0..7,16..23}; lane>=16 -> K {8..15,24..31}. Two b128 loads.
struct F16Frag { v8h lo, hi2; };

static __device__ inline F16Frag ldfrag(const _Float16* p, int hi) {
  const _Float16* q = p + (hi ? 8 : 0);
  F16Frag f;
  f.lo  = *(const v8h*)(q);
  f.hi2 = *(const v8h*)(q + 16);
  return f;
}

static __device__ inline v16h asm16(F16Frag f) {
  v16h r;
#pragma unroll
  for (int i = 0; i < 8; ++i) { r[i] = f.lo[i]; r[8 + i] = f.hi2[i]; }
  return r;
}

static __device__ inline v8f wmma16(v16h a, v16h b, v8f c) {
  return __builtin_amdgcn_wmma_f32_16x16x32_f16(false, a, false, b, (short)0, c,
                                                false, false);
}

// XOR swizzle for the 16x2048 f16 LDS tile: spreads 16B chunks across banks.
static __device__ inline int swz(int row, int col) {
  return row * 2048 + ((((col >> 3) ^ row) & 255) << 3) + (col & 7);
}

// ================= kernel 0: f32 -> f16 bulk convert ========================
__global__ __launch_bounds__(256) void k_cvt(const float* __restrict__ src,
                                             _Float16* __restrict__ dst,
                                             long long n) {
  const long long i = ((long long)blockIdx.x * 256 + threadIdx.x) * 8;
  if (i + 8 <= n) {
    v4f a = *(const v4f*)(src + i);
    v4f b = *(const v4f*)(src + i + 4);
    v8h o;
#pragma unroll
    for (int j = 0; j < 4; ++j) {
      o[j]     = (_Float16)a[j];
      o[4 + j] = (_Float16)b[j];
    }
    *(v8h*)(dst + i) = o;
  }
}

// ================= kernel 1: fused QKV projection + rotary ==================
// z = 0:Q, 1:K (rotary), 2:V (no rotary, stored transposed [h][b][dv][L]).
// All GEMM operands pre-converted to f16; K-loop software-pipelined.
__global__ __launch_bounds__(256) void k_qkv(
    const _Float16* __restrict__ xq, const _Float16* __restrict__ xk,
    const _Float16* __restrict__ xv,
    const _Float16* __restrict__ wq16, const _Float16* __restrict__ wk16,
    const _Float16* __restrict__ wv16,
    const float* __restrict__ bq, const float* __restrict__ bk,
    const float* __restrict__ bvp,
    _Float16* __restrict__ qh, _Float16* __restrict__ kh,
    _Float16* __restrict__ vt) {
  const int z = blockIdx.z;
  const _Float16* X = (z == 0) ? xq : (z == 1) ? xk : xv;
  const _Float16* W = (z == 0) ? wq16 : (z == 1) ? wk16 : wv16;
  const float* bias = (z == 0) ? bq : (z == 1) ? bk : bvp;
  _Float16* dstQK   = (z == 0) ? qh : kh;

  const int tid = threadIdx.x, lane = tid & 31, wid = tid >> 5;
  const int hi = lane >> 4, ln = lane & 15;
  const int mrow = blockIdx.x * 64 + (wid & 3) * 16;
  const int ncol = blockIdx.y * 128 + (wid >> 2) * 64;

  const _Float16* arow = X + (long long)(mrow + ln) * DM_;
  const _Float16* w0 = W + (long long)(ncol + 0 * 16 + ln) * DM_;
  const _Float16* w1 = W + (long long)(ncol + 1 * 16 + ln) * DM_;
  const _Float16* w2 = W + (long long)(ncol + 2 * 16 + ln) * DM_;
  const _Float16* w3 = W + (long long)(ncol + 3 * 16 + ln) * DM_;

  v8f acc[4] = {{}, {}, {}, {}};
  // software pipeline: fragments for step k are loaded at step k-1
  F16Frag fa = ldfrag(arow, hi);
  F16Frag f0 = ldfrag(w0, hi), f1 = ldfrag(w1, hi);
  F16Frag f2 = ldfrag(w2, hi), f3 = ldfrag(w3, hi);
  for (int k0 = 32; k0 < DM_; k0 += 32) {
    F16Frag na = ldfrag(arow + k0, hi);
    F16Frag n0 = ldfrag(w0 + k0, hi);
    F16Frag n1 = ldfrag(w1 + k0, hi);
    F16Frag n2 = ldfrag(w2 + k0, hi);
    F16Frag n3 = ldfrag(w3 + k0, hi);
    v16h af = asm16(fa);
    acc[0] = wmma16(af, asm16(f0), acc[0]);
    acc[1] = wmma16(af, asm16(f1), acc[1]);
    acc[2] = wmma16(af, asm16(f2), acc[2]);
    acc[3] = wmma16(af, asm16(f3), acc[3]);
    fa = na; f0 = n0; f1 = n1; f2 = n2; f3 = n3;
  }
  {
    v16h af = asm16(fa);
    acc[0] = wmma16(af, asm16(f0), acc[0]);
    acc[1] = wmma16(af, asm16(f1), acc[1]);
    acc[2] = wmma16(af, asm16(f2), acc[2]);
    acc[3] = wmma16(af, asm16(f3), acc[3]);
  }

  const bool doRot = (z < 2);
#pragma unroll
  for (int t = 0; t < 4; ++t) {
    const int n = ncol + t * 16 + ln;
    const float bn = bias[n];
    const float inv = __expf((float)(n >> 1) * ANGC_);
    const int h = n >> 6, d = n & 63;
#pragma unroll
    for (int r = 0; r < 8; ++r) {
      float val = acc[t][r] + bn;
      float part = __shfl_xor(val, 1, 32);  // rotary partner col (n^1)
      const int m  = mrow + hi * 8 + r;
      const int bb = m >> 11;
      const int pl = m & (L_ - 1);
      if (doRot) {
        float sa, ca;
        __sincosf((float)(pl + 1) * inv, &sa, &ca);
        val = ((n & 1) == 0) ? (val * ca - part * sa) : (val * ca + part * sa);
      }
      if (z == 2) {
        vt[((long long)(h * B_ + bb) * DV_ + d) * L_ + pl] = (_Float16)val;
      } else {
        dstQK[(((long long)(h * B_ + bb) * L_) + pl) * DK_ + d] = (_Float16)val;
      }
    }
  }
}

// ======== kernel 2: fused scores + mask + softmax + attn@V ==================
__global__ __launch_bounds__(256) void k_attn(
    const _Float16* __restrict__ qh, const _Float16* __restrict__ kh,
    const _Float16* __restrict__ vt, const unsigned char* __restrict__ mask,
    float* __restrict__ attn, _Float16* __restrict__ ctx16) {
  __shared__ _Float16 SM[16 * 2048];  // 64 KB, aliased 3 ways
  const int tid = threadIdx.x, lane = tid & 31, w = tid >> 5;
  const int hi = lane >> 4, ln = lane & 15;
  const int hb = blockIdx.y;
  const int h = hb >> 1, b = hb & 1;
  const int q0 = blockIdx.x * 16;

  const _Float16* qb = qh + (long long)hb * L_ * DK_;
  const _Float16* kb = kh + (long long)hb * L_ * DK_;

  v16h a0 = asm16(ldfrag(qb + (long long)(q0 + ln) * DK_, hi));
  v16h a1 = asm16(ldfrag(qb + (long long)(q0 + ln) * DK_ + 32, hi));

  // ---- stage 1: logits for 16 x 256 slab, in registers, pipelined ----
  v8f s[16];
  const v8f zero = {};
  const _Float16* kbase = kb + (long long)(w * 256 + ln) * DK_;
  F16Frag b0 = ldfrag(kbase, hi);
  F16Frag b1 = ldfrag(kbase + 32, hi);
#pragma unroll
  for (int t = 0; t < 15; ++t) {
    F16Frag n0 = ldfrag(kbase + (long long)(t + 1) * 16 * DK_, hi);
    F16Frag n1 = ldfrag(kbase + (long long)(t + 1) * 16 * DK_ + 32, hi);
    s[t] = wmma16(a0, asm16(b0), zero);
    s[t] = wmma16(a1, asm16(b1), s[t]);
    b0 = n0; b1 = n1;
  }
  s[15] = wmma16(a0, asm16(b0), zero);
  s[15] = wmma16(a1, asm16(b1), s[15]);

  const float NEG = -__builtin_inff();
#pragma unroll
  for (int t = 0; t < 16; ++t) {
    const int n = w * 256 + t * 16 + ln;
    const unsigned char* mp = mask + (long long)b * L_ * L_ + n;
#pragma unroll
    for (int r = 0; r < 8; ++r) {
      const int qq = q0 + hi * 8 + r;
      s[t][r] = mp[(long long)qq * L_] ? NEG : s[t][r] * 0.125f;
    }
  }

  // ---- row max: lane-local -> shfl across 16 lanes -> LDS across waves ----
  float* stats = (float*)SM;  // [8 waves][16 rows] (SM not yet in use)
  float rmax[8];
#pragma unroll
  for (int r = 0; r < 8; ++r) {
    float m = s[0][r];
#pragma unroll
    for (int t = 1; t < 16; ++t) m = fmaxf(m, s[t][r]);
#pragma unroll
    for (int d = 1; d < 16; d <<= 1) m = fmaxf(m, __shfl_xor(m, d, 32));
    rmax[r] = m;
  }
  if (ln == 0) {
#pragma unroll
    for (int r = 0; r < 8; ++r) stats[w * 16 + hi * 8 + r] = rmax[r];
  }
  __syncthreads();
#pragma unroll
  for (int r = 0; r < 8; ++r) {
    float m = stats[hi * 8 + r];
#pragma unroll
    for (int ww = 1; ww < 8; ++ww) m = fmaxf(m, stats[ww * 16 + hi * 8 + r]);
    rmax[r] = m;
  }
  __syncthreads();  // all stats reads done before region is rewritten

  // ---- exp + row sum ----
  float rsum[8];
#pragma unroll
  for (int r = 0; r < 8; ++r) {
    const bool dead = (rmax[r] == NEG);  // fully masked row -> all zeros
    float a = 0.f;
#pragma unroll
    for (int t = 0; t < 16; ++t) {
      float e = dead ? 0.f : __expf(s[t][r] - rmax[r]);
      s[t][r] = e;
      a += e;
    }
#pragma unroll
    for (int d = 1; d < 16; d <<= 1) a += __shfl_xor(a, d, 32);
    rsum[r] = a;
  }
  if (ln == 0) {
#pragma unroll
    for (int r = 0; r < 8; ++r) stats[w * 16 + hi * 8 + r] = rsum[r];
  }
  __syncthreads();
#pragma unroll
  for (int r = 0; r < 8; ++r) {
    float sv = 0.f;
#pragma unroll
    for (int ww = 0; ww < 8; ++ww) sv += stats[ww * 16 + hi * 8 + r];
    rsum[r] = (sv > 0.f) ? 1.f / sv : 0.f;
  }
  __syncthreads();  // before SM is reused for f16 probabilities

  // ---- normalize: write attn (f32, global) + staged f16 tile (LDS) ----
#pragma unroll
  for (int t = 0; t < 16; ++t) {
    const int n = w * 256 + t * 16 + ln;
#pragma unroll
    for (int r = 0; r < 8; ++r) {
      const int row = hi * 8 + r;
      const float p = s[t][r] * rsum[r];
      attn[((long long)hb * L_ + q0 + row) * L_ + n] = p;
      SM[swz(row, n)] = (_Float16)p;
    }
  }
  __syncthreads();

  // ---- stage 2: ctx_partial = P[:, slab] @ V[slab, :], pipelined ----
  const _Float16* vb = vt + (long long)hb * DV_ * L_;
  const _Float16* v0 = vb + (long long)(0 * 16 + ln) * L_;
  const _Float16* v1 = vb + (long long)(1 * 16 + ln) * L_;
  const _Float16* v2 = vb + (long long)(2 * 16 + ln) * L_;
  const _Float16* v3 = vb + (long long)(3 * 16 + ln) * L_;
  const int kbeg = w * 256;

  v8f c4[4] = {{}, {}, {}, {}};
  F16Frag g0 = ldfrag(v0 + kbeg, hi), g1 = ldfrag(v1 + kbeg, hi);
  F16Frag g2 = ldfrag(v2 + kbeg, hi), g3 = ldfrag(v3 + kbeg, hi);
#pragma unroll
  for (int kk = 0; kk < 8; ++kk) {
    const int k0 = kbeg + kk * 32;
    F16Frag n0, n1, n2, n3;
    if (kk < 7) {
      n0 = ldfrag(v0 + k0 + 32, hi);
      n1 = ldfrag(v1 + k0 + 32, hi);
      n2 = ldfrag(v2 + k0 + 32, hi);
      n3 = ldfrag(v3 + k0 + 32, hi);
    }
    const int h0 = k0 + (hi ? 8 : 0);
    v8h xlo = *(const v8h*)(SM + swz(ln, h0));       // ds_load_b128
    v8h xhi = *(const v8h*)(SM + swz(ln, h0 + 16));
    v16h af;
#pragma unroll
    for (int i = 0; i < 8; ++i) { af[i] = xlo[i]; af[8 + i] = xhi[i]; }
    c4[0] = wmma16(af, asm16(g0), c4[0]);
    c4[1] = wmma16(af, asm16(g1), c4[1]);
    c4[2] = wmma16(af, asm16(g2), c4[2]);
    c4[3] = wmma16(af, asm16(g3), c4[3]);
    if (kk < 7) { g0 = n0; g1 = n1; g2 = n2; g3 = n3; }
  }
  __syncthreads();  // all LDS prob reads done -> reuse SM as f32 partials

  float* CF = (float*)SM;  // [8 waves][16 rows][64 cols] = 32 KB
#pragma unroll
  for (int t = 0; t < 4; ++t) {
#pragma unroll
    for (int r = 0; r < 8; ++r)
      CF[w * 1024 + (hi * 8 + r) * 64 + t * 16 + ln] = c4[t][r];
  }
  __syncthreads();

  // cross-wave sum + store ctx as f16 (GEMM A-operand for k_fc)
  {
    const int row = tid >> 4, col = (tid & 15) * 4;
    v4f a = *(const v4f*)(CF + row * 64 + col);
#pragma unroll
    for (int ww = 1; ww < 8; ++ww)
      a += *(const v4f*)(CF + ww * 1024 + row * 64 + col);
    v4h o;
#pragma unroll
    for (int j = 0; j < 4; ++j) o[j] = (_Float16)a[j];
    *(v4h*)(ctx16 + (long long)(b * L_ + q0 + row) * NP_ + h * DV_ + col) = o;
  }
}

// ================= kernel 3: y' = ctx @ Wfc^T + b + residual ================
__global__ __launch_bounds__(256) void k_fc(
    const _Float16* __restrict__ ctx16, const _Float16* __restrict__ wfc16,
    const float* __restrict__ bfc, const float* __restrict__ resid,
    float* __restrict__ y) {
  const int tid = threadIdx.x, lane = tid & 31, wid = tid >> 5;
  const int hi = lane >> 4, ln = lane & 15;
  const int mrow = blockIdx.x * 64 + (wid & 3) * 16;
  const int ncol = blockIdx.y * 128 + (wid >> 2) * 64;

  const _Float16* arow = ctx16 + (long long)(mrow + ln) * NP_;
  const _Float16* w0 = wfc16 + (long long)(ncol + 0 * 16 + ln) * NP_;
  const _Float16* w1 = wfc16 + (long long)(ncol + 1 * 16 + ln) * NP_;
  const _Float16* w2 = wfc16 + (long long)(ncol + 2 * 16 + ln) * NP_;
  const _Float16* w3 = wfc16 + (long long)(ncol + 3 * 16 + ln) * NP_;

  v8f acc[4] = {{}, {}, {}, {}};
  F16Frag fa = ldfrag(arow, hi);
  F16Frag f0 = ldfrag(w0, hi), f1 = ldfrag(w1, hi);
  F16Frag f2 = ldfrag(w2, hi), f3 = ldfrag(w3, hi);
  for (int k0 = 32; k0 < NP_; k0 += 32) {
    F16Frag na = ldfrag(arow + k0, hi);
    F16Frag n0 = ldfrag(w0 + k0, hi);
    F16Frag n1 = ldfrag(w1 + k0, hi);
    F16Frag n2 = ldfrag(w2 + k0, hi);
    F16Frag n3 = ldfrag(w3 + k0, hi);
    v16h af = asm16(fa);
    acc[0] = wmma16(af, asm16(f0), acc[0]);
    acc[1] = wmma16(af, asm16(f1), acc[1]);
    acc[2] = wmma16(af, asm16(f2), acc[2]);
    acc[3] = wmma16(af, asm16(f3), acc[3]);
    fa = na; f0 = n0; f1 = n1; f2 = n2; f3 = n3;
  }
  {
    v16h af = asm16(fa);
    acc[0] = wmma16(af, asm16(f0), acc[0]);
    acc[1] = wmma16(af, asm16(f1), acc[1]);
    acc[2] = wmma16(af, asm16(f2), acc[2]);
    acc[3] = wmma16(af, asm16(f3), acc[3]);
  }
#pragma unroll
  for (int t = 0; t < 4; ++t) {
    const int n = ncol + t * 16 + ln;
    const float bn = bfc[n];
#pragma unroll
    for (int r = 0; r < 8; ++r) {
      const int m = mrow + hi * 8 + r;
      y[(long long)m * DM_ + n] = acc[t][r] + bn + resid[(long long)m * DM_ + n];
    }
  }
}

// ================= kernel 4: in-place LayerNorm over DM =====================
__global__ __launch_bounds__(256) void k_ln(float* __restrict__ y,
                                            const float* __restrict__ gamma,
                                            const float* __restrict__ beta) {
  __shared__ float s1[256], s2[256];
  const int tid = threadIdx.x;
  float* p = y + (long long)blockIdx.x * DM_;
  v4f x = *(v4f*)(p + tid * 4);
  s1[tid] = x[0] + x[1] + x[2] + x[3];
  s2[tid] = x[0]*x[0] + x[1]*x[1] + x[2]*x[2] + x[3]*x[3];
  __syncthreads();
  for (int s = 128; s > 0; s >>= 1) {
    if (tid < s) { s1[tid] += s1[tid + s]; s2[tid] += s2[tid + s]; }
    __syncthreads();
  }
  const float mu  = s1[0] * (1.0f / DM_);
  const float var = s2[0] * (1.0f / DM_) - mu * mu;
  const float inv = rsqrtf(var + 1e-5f);
#pragma unroll
  for (int i = 0; i < 4; ++i) {
    const int n = tid * 4 + i;
    x[i] = (x[i] - mu) * inv * gamma[n] + beta[n];
  }
  *(v4f*)(p + tid * 4) = x;
}

// ============================ launch =========================================
extern "C" void kernel_launch(void* const* d_in, const int* in_sizes, int n_in,
                              void* d_out, int out_size, void* d_ws,
                              size_t ws_size, hipStream_t stream) {
  const float* qin  = (const float*)d_in[0];
  const float* kin  = (const float*)d_in[1];
  const float* vin  = (const float*)d_in[2];
  const unsigned char* mask = (const unsigned char*)d_in[3];  // bool mask
  const float* wq  = (const float*)d_in[4];
  const float* bq  = (const float*)d_in[5];
  const float* wk  = (const float*)d_in[6];
  const float* bk  = (const float*)d_in[7];
  const float* wv  = (const float*)d_in[8];
  const float* bv  = (const float*)d_in[9];
  const float* wfc = (const float*)d_in[10];
  const float* bfc = (const float*)d_in[11];
  const float* gamma = (const float*)d_in[12];
  const float* beta  = (const float*)d_in[13];

  float* y    = (float*)d_out;                 // (B,L,DM)
  float* attn = y + (long long)B_ * L_ * DM_;  // (H*B,L,L)

  const long long NX = (long long)B_ * L_ * DM_;   // 4 Mi elems
  const long long NW = (long long)NP_ * DM_;       // 1 Mi elems

  _Float16* qh    = (_Float16*)d_ws;       // projected Q, head-major
  _Float16* kh    = qh + NX;
  _Float16* vt    = kh + NX;               // projected V, transposed
  _Float16* ctx16 = vt + NX;
  _Float16* xq    = ctx16 + NX;            // converted inputs
  _Float16* xk    = xq + NX;
  _Float16* xv    = xk + NX;
  _Float16* wq16  = xv + NX;               // converted weights
  _Float16* wk16  = wq16 + NW;
  _Float16* wv16  = wk16 + NW;
  _Float16* wfc16 = wv16 + NW;

  // bulk f32->f16 conversions (bandwidth-trivial)
  k_cvt<<<dim3((unsigned)(NX / 2048)), 256, 0, stream>>>(qin, xq, NX);
  k_cvt<<<dim3((unsigned)(NX / 2048)), 256, 0, stream>>>(kin, xk, NX);
  k_cvt<<<dim3((unsigned)(NX / 2048)), 256, 0, stream>>>(vin, xv, NX);
  k_cvt<<<dim3((unsigned)(NW / 2048)), 256, 0, stream>>>(wq,  wq16, NW);
  k_cvt<<<dim3((unsigned)(NW / 2048)), 256, 0, stream>>>(wk,  wk16, NW);
  k_cvt<<<dim3((unsigned)(NW / 2048)), 256, 0, stream>>>(wv,  wv16, NW);
  k_cvt<<<dim3((unsigned)(NW / 2048)), 256, 0, stream>>>(wfc, wfc16, NW);

  k_qkv <<<dim3(64, 8, 3), 256, 0, stream>>>(xq, xk, xv, wq16, wk16, wv16,
                                             bq, bk, bv, qh, kh, vt);
  k_attn<<<dim3(128, 32),  256, 0, stream>>>(qh, kh, vt, mask, attn, ctx16);
  k_fc  <<<dim3(64, 8),    256, 0, stream>>>(ctx16, wfc16, bfc, qin, y);
  k_ln  <<<dim3(4096),     256, 0, stream>>>(y, gamma, beta);
}
